// Attention_kv_12979391169262
// MI455X (gfx1250) — matrix operand order repaired and checked by reading, not compiled, not run on hardware
//
#include <hip/hip_runtime.h>
#include <hip/hip_bf16.h>
#include <stdint.h>

typedef __attribute__((ext_vector_type(16))) __bf16 v16bf;
typedef __attribute__((ext_vector_type(8)))  float  v8f;
typedef __attribute__((ext_vector_type(4))) unsigned u32x4;
typedef __attribute__((ext_vector_type(4))) int      i32x4;
typedef __attribute__((ext_vector_type(8))) int      i32x8;

#if __has_builtin(__builtin_amdgcn_tensor_load_to_lds)
#define USE_TDM 1
#endif

#define DEV static __device__ __forceinline__

DEV unsigned short f2bf(float f) {
  union { float f; unsigned u; } in; in.f = f;
  unsigned u = in.u;
  unsigned r = (u + 0x7FFFu + ((u >> 16) & 1u)) >> 16;  // RNE
  return (unsigned short)r;
}
DEV float bf2f(unsigned short h) {
  union { unsigned u; float f; } out; out.u = ((unsigned)h) << 16;
  return out.f;
}

union Frag16 { uint4 q[2]; v16bf v; };

// ---------------------------------------------------------------------------
// f32 -> bf16 elementwise convert
// ---------------------------------------------------------------------------
__global__ void k_cvt_bf16(const float* __restrict__ in,
                           unsigned short* __restrict__ out, int n) {
  int i = blockIdx.x * (blockDim.x * 4) + threadIdx.x;
  #pragma unroll
  for (int j = 0; j < 4; ++j) {
    int idx = i + j * 256;
    if (idx < n) out[idx] = f2bf(in[idx]);
  }
}

// ---------------------------------------------------------------------------
// Weight transpose: W f32 [K x N] -> Wt bf16 [N x K] (B-operand layout)
// ---------------------------------------------------------------------------
__global__ void k_transpose_w(const float* __restrict__ in,
                              unsigned short* __restrict__ out, int K, int N) {
  long long idx = (long long)blockIdx.x * blockDim.x + threadIdx.x;
  long long total = (long long)K * N;
  if (idx >= total) return;
  long long n = idx / K, k = idx % K;
  out[idx] = f2bf(in[k * (long long)N + n]);
}

// ---------------------------------------------------------------------------
// bf16 strided transpose (V -> V^T), 32x32 tiles staged through LDS.
// Preferred path: Tensor Data Mover (tensor_load_to_lds + s_wait_tensorcnt):
// one wave per WG issues a 2D-tile D# (data_size=2B, tile 32x32, dim0_stride =
// row stride), TDM packs the tile densely into LDS. Fallback: CDNA5 async
// global->LDS loads (ASYNCcnt).
// in:  [R x C]  row stride inStride   ->  out: [C x R] row stride outStride
// grid: (C/32, R/32, batch), block 256
// ---------------------------------------------------------------------------
__global__ void k_transpose_bf16(const unsigned short* __restrict__ in,
                                 long long inStride, long long inBatch,
                                 unsigned short* __restrict__ out,
                                 long long outStride, long long outBatch) {
#ifdef USE_TDM
  __shared__ __align__(16) unsigned short tile[32][32];  // dense: TDM packs rows
#else
  __shared__ __align__(16) unsigned short tile[32][40];  // 80B rows: 16B aligned
#endif
  const int tid = threadIdx.x;
  const unsigned short* ib = in + (long long)blockIdx.z * inBatch;
  unsigned short* ob = out + (long long)blockIdx.z * outBatch;
  const long long r0 = (long long)blockIdx.y * 32;
  const long long c0 = (long long)blockIdx.x * 32;

#ifdef USE_TDM
  if (tid < 32) {  // first wave issues the DMA (TDM is wave-level, EXEC ignored)
    const unsigned short* gp = ib + r0 * inStride + c0;
    unsigned long long ga  = (unsigned long long)(uintptr_t)gp;   // byte address
    unsigned           lds = (unsigned)(uintptr_t)(&tile[0][0]);  // LDS offset
    unsigned long long str = (unsigned long long)inStride;        // 2B units
    // D# group0: [1:0]count=1 | [63:32]lds_addr | [120:64]global_addr | type=2
    u32x4 g0 = { 1u,
                 lds,
                 (unsigned)(ga & 0xFFFFFFFFull),
                 (unsigned)((ga >> 32) & 0x1FFFFFFull) | (2u << 30) };
    // D# group1: data_size=2B; tensor_dim0/1 large (tile fully in-bounds);
    // tile_dim0=32, tile_dim1=32, tile_dim2=0; dim0_stride = row stride.
    i32x8 g1 = { (int)(1u << 16),            // [17:16] data_size=1 -> 2 bytes
                 (int)0xFFFF0000u,           // [63:48]  tensor_dim0 lo16
                 (int)0xFFFF000Fu,           // dim0 hi16 | dim1 lo16
                 (int)((32u << 16) | 0xFu),  // dim1 hi16 | tile_dim0=32
                 (int)32,                    // tile_dim1=32, tile_dim2=0
                 (int)(str & 0xFFFFFFFFull), // dim0_stride[31:0]
                 (int)((str >> 32) & 0xFFFFull),
                 0 };
    i32x4 z4 = { 0, 0, 0, 0 };
#if defined(__clang_major__) && __clang_major__ >= 23
    i32x8 z8 = { 0, 0, 0, 0, 0, 0, 0, 0 };
    __builtin_amdgcn_tensor_load_to_lds(g0, g1, z4, z4, z8, 0);
#else
    __builtin_amdgcn_tensor_load_to_lds(g0, g1, z4, z4, 0);
#endif
#if __has_builtin(__builtin_amdgcn_s_wait_tensorcnt)
    __builtin_amdgcn_s_wait_tensorcnt(0);
#else
    asm volatile("s_wait_tensorcnt 0x0" ::: "memory");
#endif
  }
  __syncthreads();
#else
  if (tid < 128) {  // 128 lanes x 16B = 32x32 bf16 tile
    const int row = tid >> 2, chunk = tid & 3;
    const unsigned short* gp = ib + (r0 + row) * inStride + c0 + chunk * 8;
    unsigned ldsoff = (unsigned)(uintptr_t)(&tile[row][chunk * 8]);
    asm volatile("global_load_async_to_lds_b128 %0, %1, off"
                 :: "v"(ldsoff), "v"(gp) : "memory");
  }
  asm volatile("s_wait_asynccnt 0x0" ::: "memory");
  __syncthreads();
#endif

  const int oc  = tid >> 3;        // output row (= input col) 0..31
  const int orr = (tid & 7) * 4;   // input row base
  #pragma unroll
  for (int i = 0; i < 4; ++i)
    ob[(c0 + oc) * outStride + (r0 + orr + i)] = tile[orr + i][oc];
}

// ---------------------------------------------------------------------------
// bf16 WMMA GEMM:  C[M,N] = alpha * A[M,K] @ Bt[N,K]^T + bias[N]
//   A  : bf16 row-major, row stride aStride, batch stride aBatch
//   Bt : bf16, N rows of K contiguous values (i.e. B^T row-major)
//   C  : f32 (cF32) or bf16
// Workgroup = 8 waves, tile 64(M) x 128(N); wave w owns 64x16 (4 C-frags),
// B fragment reused across the 4 M-subtiles.
// grid: (N/128, M/64, batch), block 256.  M%64==0, N%128==0, K%32==0.
// ---------------------------------------------------------------------------
DEV void store_tile(const v8f& acc, long long row, long long col,
                    long long cStride, float alpha, float bv, int cF32,
                    float* cf, unsigned short* ch) {
  #pragma unroll
  for (int r = 0; r < 8; ++r) {
    float v = alpha * acc[r] + bv;
    if (cF32) cf[(row + r) * cStride + col] = v;
    else      ch[(row + r) * cStride + col] = f2bf(v);
  }
}

__global__ void k_gemm_wmma(const unsigned short* __restrict__ A,
                            long long aStride, long long aBatch,
                            const unsigned short* __restrict__ Bt,
                            long long bStride, long long bBatch,
                            const float* __restrict__ bias,
                            void* __restrict__ C,
                            long long cStride, long long cBatch, int cF32,
                            int Kdim, float alpha) {
  const int w    = threadIdx.x >> 5;
  const int lane = threadIdx.x & 31;
  const int lhi  = lane >> 4;   // half-wave select
  const int llo  = lane & 15;
  const long long tileM = (long long)blockIdx.y * 64;
  const long long tileN = (long long)blockIdx.x * 128 + (long long)w * 16;
  const unsigned short* Ab = A  + (long long)blockIdx.z * aBatch;
  const unsigned short* Bb = Bt + (long long)blockIdx.z * bBatch;

  v8f acc0 = {}, acc1 = {}, acc2 = {}, acc3 = {};

  // A frag per lane: row = llo; K chunks at lhi*8 and 16+lhi*8
  const unsigned short* pa0 = Ab + (tileM +  0 + llo) * aStride + lhi * 8;
  const unsigned short* pa1 = Ab + (tileM + 16 + llo) * aStride + lhi * 8;
  const unsigned short* pa2 = Ab + (tileM + 32 + llo) * aStride + lhi * 8;
  const unsigned short* pa3 = Ab + (tileM + 48 + llo) * aStride + lhi * 8;
  // B frag per lane: N-row = llo; K chunks at lhi*8 and 16+lhi*8
  const unsigned short* pb  = Bb + (tileN + llo) * bStride + lhi * 8;

  for (int kk = 0; kk < Kdim; kk += 32) {
    Frag16 b;
    b.q[0] = *(const uint4*)(pb + kk);
    b.q[1] = *(const uint4*)(pb + kk + 16);
    Frag16 a0, a1, a2, a3;
    a0.q[0] = *(const uint4*)(pa0 + kk); a0.q[1] = *(const uint4*)(pa0 + kk + 16);
    a1.q[0] = *(const uint4*)(pa1 + kk); a1.q[1] = *(const uint4*)(pa1 + kk + 16);
    a2.q[0] = *(const uint4*)(pa2 + kk); a2.q[1] = *(const uint4*)(pa2 + kk + 16);
    a3.q[0] = *(const uint4*)(pa3 + kk); a3.q[1] = *(const uint4*)(pa3 + kk + 16);
    acc0 = __builtin_amdgcn_wmma_f32_16x16x32_bf16(false, a0.v, false, b.v, (short)0, acc0, false, false);
    acc1 = __builtin_amdgcn_wmma_f32_16x16x32_bf16(false, a1.v, false, b.v, (short)0, acc1, false, false);
    acc2 = __builtin_amdgcn_wmma_f32_16x16x32_bf16(false, a2.v, false, b.v, (short)0, acc2, false, false);
    acc3 = __builtin_amdgcn_wmma_f32_16x16x32_bf16(false, a3.v, false, b.v, (short)0, acc3, false, false);
  }

  const long long col = tileN + llo;
  const float bv = bias ? bias[col] : 0.0f;
  float* cf          = (float*)C          + (long long)blockIdx.z * cBatch;
  unsigned short* ch = (unsigned short*)C + (long long)blockIdx.z * cBatch;
  store_tile(acc0, tileM +  0 + lhi * 8, col, cStride, alpha, bv, cF32, cf, ch);
  store_tile(acc1, tileM + 16 + lhi * 8, col, cStride, alpha, bv, cF32, cf, ch);
  store_tile(acc2, tileM + 32 + lhi * 8, col, cStride, alpha, bv, cF32, cf, ch);
  store_tile(acc3, tileM + 48 + lhi * 8, col, cStride, alpha, bv, cF32, cf, ch);
}

// ---------------------------------------------------------------------------
// Masked softmax, in place over bf16 scores. One 256-thread block per row
// (row length 1024, 4 elements/thread). pair mask = mask[m]*mask[n] != 0,
// invalid -> -10000 (matches reference). grid: (M, B).
// ---------------------------------------------------------------------------
__global__ void k_softmax_mask(unsigned short* __restrict__ S,
                               const float* __restrict__ mask, int Mlen) {
  const int b = blockIdx.y, m = blockIdx.x;
  unsigned short* row = S + ((long long)b * Mlen + m) * (long long)Mlen;
  const float* mk = mask + (long long)b * Mlen;
  const float mrow = mk[m];
  __shared__ float red[8];
  const int tid = threadIdx.x, lane = tid & 31, wid = tid >> 5;

  float vals[4];
  float vmax = -3.0e38f;
  #pragma unroll
  for (int j = 0; j < 4; ++j) {
    int n = j * 256 + tid;
    float s = bf2f(row[n]);
    s = (mrow != 0.0f && mk[n] != 0.0f) ? s : -10000.0f;
    vals[j] = s;
    vmax = fmaxf(vmax, s);
  }
  #pragma unroll
  for (int o = 16; o > 0; o >>= 1) vmax = fmaxf(vmax, __shfl_xor(vmax, o));
  if (lane == 0) red[wid] = vmax;
  __syncthreads();
  float gmax = red[0];
  #pragma unroll
  for (int j = 1; j < 8; ++j) gmax = fmaxf(gmax, red[j]);

  float lsum = 0.0f;
  #pragma unroll
  for (int j = 0; j < 4; ++j) { vals[j] = __expf(vals[j] - gmax); lsum += vals[j]; }
  #pragma unroll
  for (int o = 16; o > 0; o >>= 1) lsum += __shfl_xor(lsum, o);
  __syncthreads();
  if (lane == 0) red[wid] = lsum;
  __syncthreads();
  float gsum = 0.0f;
  #pragma unroll
  for (int j = 0; j < 8; ++j) gsum += red[j];
  const float inv = 1.0f / gsum;
  #pragma unroll
  for (int j = 0; j < 4; ++j) row[j * 256 + tid] = f2bf(vals[j] * inv);
}

// ---------------------------------------------------------------------------
extern "C" void kernel_launch(void* const* d_in, const int* in_sizes, int n_in,
                              void* d_out, int out_size, void* d_ws, size_t ws_size,
                              hipStream_t stream) {
  (void)in_sizes; (void)n_in; (void)out_size; (void)ws_size;
  const int Bb = 8, Mm = 1024, Cc = 768;
  const long long R = (long long)Bb * Mm;                 // 8192
  const float scale = 0.036084391824351615f;              // 768^-0.5

  const float* layout_x = (const float*)d_in[0];
  const float* text_x   = (const float*)d_in[1];
  const float* mask     = (const float*)d_in[2];
  const float* Wqkv     = (const float*)d_in[3];
  const float* bqkv     = (const float*)d_in[4];
  const float* Wq       = (const float*)d_in[5];
  const float* bq       = (const float*)d_in[6];
  const float* Wkv      = (const float*)d_in[7];
  const float* bkv      = (const float*)d_in[8];
  const float* Wffn     = (const float*)d_in[9];
  const float* bffn     = (const float*)d_in[10];
  float* out = (float*)d_out;

  char* ws = (char*)d_ws;
  size_t off = 0;
  auto alloc16 = [&](size_t elems) -> unsigned short* {
    unsigned short* p = (unsigned short*)(ws + off);
    off = (off + elems * 2 + 255) & ~(size_t)255;
    return p;
  };
  unsigned short* XL    = alloc16((size_t)R * Cc);
  unsigned short* XT    = alloc16((size_t)R * Cc);
  unsigned short* WQKVT = alloc16((size_t)3 * Cc * Cc);
  unsigned short* WQT   = alloc16((size_t)Cc * Cc);
  unsigned short* WKVT  = alloc16((size_t)2 * Cc * Cc);
  unsigned short* WFFNT = alloc16((size_t)Cc * Cc);
  unsigned short* QKV   = alloc16((size_t)R * 3 * Cc);    // reused as KV (layer 2)
  unsigned short* S     = alloc16((size_t)Bb * Mm * Mm);  // scores/probs, both layers
  unsigned short* VT    = alloc16((size_t)Bb * Cc * Mm);  // V^T, both layers
  unsigned short* LO    = alloc16((size_t)R * Cc);        // self-attn output
  unsigned short* CQ = XL;   // layout_x dead after QKV gemm
  unsigned short* MG = XT;   // text_x dead after KV gemm
  unsigned short* KV = QKV;  // qkv dead after layer-1 attention

  // --- prep: bf16 conversions + weight transposes ---
  {
    int n = (int)(R * Cc);
    int grid = (n + 1023) / 1024;
    k_cvt_bf16<<<grid, 256, 0, stream>>>(layout_x, XL, n);
    k_cvt_bf16<<<grid, 256, 0, stream>>>(text_x,   XT, n);
  }
  auto launchWT = [&](const float* Win, unsigned short* Wt, int K, int N) {
    long long total = (long long)K * N;
    k_transpose_w<<<(int)((total + 255) / 256), 256, 0, stream>>>(Win, Wt, K, N);
  };
  launchWT(Wqkv, WQKVT, Cc, 3 * Cc);
  launchWT(Wq,   WQT,   Cc, Cc);
  launchWT(Wkv,  WKVT,  Cc, 2 * Cc);
  launchWT(Wffn, WFFNT, Cc, Cc);

  auto gemm = [&](const unsigned short* A, long long aS, long long aB,
                  const unsigned short* Bt, long long bS, long long bB,
                  const float* bias, void* Co, long long cS, long long cB,
                  int cF32, int Mr, int Nc, int Kd, int batch, float alpha) {
    dim3 g((unsigned)(Nc / 128), (unsigned)(Mr / 64), (unsigned)batch);
    k_gemm_wmma<<<g, 256, 0, stream>>>(A, aS, aB, Bt, bS, bB, bias,
                                       Co, cS, cB, cF32, Kd, alpha);
  };

  // --- layer 1: masked self-attention on layout_x ---
  gemm(XL, Cc, 0, WQKVT, Cc, 0, bqkv, QKV, 3 * Cc, 0, 0, (int)R, 3 * Cc, Cc, 1, 1.0f);
  // S = scale * Q @ K^T   (K rows are already K-contiguous -> direct B operand)
  gemm(QKV, 3 * Cc, (long long)Mm * 3 * Cc, QKV + Cc, 3 * Cc, (long long)Mm * 3 * Cc,
       nullptr, S, Mm, (long long)Mm * Mm, 0, Mm, Mm, Cc, Bb, scale);
  k_softmax_mask<<<dim3(Mm, Bb), 256, 0, stream>>>(S, mask, Mm);
  k_transpose_bf16<<<dim3(Cc / 32, Mm / 32, Bb), 256, 0, stream>>>(
      QKV + 2 * Cc, 3 * Cc, (long long)Mm * 3 * Cc, VT, Mm, (long long)Cc * Mm);
  gemm(S, Mm, (long long)Mm * Mm, VT, Mm, (long long)Cc * Mm,
       nullptr, LO, Cc, (long long)Mm * Cc, 0, Mm, Cc, Mm, Bb, 1.0f);

  // --- layer 2: masked cross-attention (q from LO, k/v from text) ---
  gemm(XT, Cc, 0, WKVT, Cc, 0, bkv, KV, 2 * Cc, 0, 0, (int)R, 2 * Cc, Cc, 1, 1.0f);
  gemm(LO, Cc, 0, WQT,  Cc, 0, bq,  CQ, Cc,     0, 0, (int)R, Cc,     Cc, 1, 1.0f);
  gemm(CQ, Cc, (long long)Mm * Cc, KV, 2 * Cc, (long long)Mm * 2 * Cc,
       nullptr, S, Mm, (long long)Mm * Mm, 0, Mm, Mm, Cc, Bb, scale);
  k_softmax_mask<<<dim3(Mm, Bb), 256, 0, stream>>>(S, mask, Mm);
  k_transpose_bf16<<<dim3(Cc / 32, Mm / 32, Bb), 256, 0, stream>>>(
      KV + Cc, 2 * Cc, (long long)Mm * 2 * Cc, VT, Mm, (long long)Cc * Mm);
  gemm(S, Mm, (long long)Mm * Mm, VT, Mm, (long long)Cc * Mm,
       nullptr, MG, Cc, (long long)Mm * Cc, 0, Mm, Cc, Mm, Bb, 1.0f);

  // --- final projection -> f32 output ---
  gemm(MG, Cc, 0, WFFNT, Cc, 0, bffn, out, Cc, 0, 1, (int)R, Cc, Cc, 1, 1.0f);
}
